// RGCNEncoder_48722109006433
// MI455X (gfx1250) — matrix operand order repaired and checked by reading, not compiled, not run on hardware
//
#include <hip/hip_runtime.h>
#include <math.h>

#define N_ENTITY 31162
#define HID      768
#define N_REL    24
#define N_BASES  8
#define N_EDGES  600000
#define PAD_ID   31161
#define BATCH    256
#define SEQ      50
#define NPAD     31168                     // N_ENTITY padded to multiple of 64
#define KTOT     (HID * (1 + N_BASES))     // 6912 = [root | 8 bases]
#define MB2      (BATCH * SEQ)             // 12800 attention rows

typedef unsigned short u16;
typedef __attribute__((ext_vector_type(16))) __bf16        v16bf;
typedef __attribute__((ext_vector_type(8)))  float         v8f;
typedef __attribute__((ext_vector_type(4)))  unsigned int  u32x4;

union Frag32B { v16bf v; u32x4 q[2]; };   // 32 bytes = one WMMA bf16 A/B fragment per lane

__device__ __forceinline__ u16 f2bf(float f) {
  unsigned u = __float_as_uint(f);
  u += 0x7FFFu + ((u >> 16) & 1u);        // round-to-nearest-even
  return (u16)(u >> 16);
}

// CDNA5 async DMA: global -> LDS, 16B per lane, tracked by ASYNCcnt.
// LDS VGPR operand = low 32 bits of the generic shared-memory pointer
// (AS3->flat casts are materialized as {aperture_hi, lds_offset32}).
__device__ __forceinline__ void async_ld_b128(const u16* gsrc, const u16* ldst) {
  unsigned loff = (unsigned)(unsigned long long)(uintptr_t)ldst;
  asm volatile("global_load_async_to_lds_b128 %0, %1, off"
               :: "v"(loff), "v"(gsrc) : "memory");
}
__device__ __forceinline__ void wait_async0() {
  asm volatile("s_wait_asynccnt 0x0" ::: "memory");
}

// ---------------------------------------------------------------- zero fill
__global__ void k_zero_f32(float* __restrict__ p, size_t n4) {
  size_t i = (size_t)blockIdx.x * blockDim.x + threadIdx.x;
  size_t stride = (size_t)gridDim.x * blockDim.x;
  float4 z = make_float4(0.f, 0.f, 0.f, 0.f);
  for (; i < n4; i += stride) ((float4*)p)[i] = z;
}

// ---------------------------------------------------------------- per-(dst,rel) degree
__global__ void k_deg(const int* __restrict__ ei, const int* __restrict__ et,
                      float* __restrict__ deg) {
  int e = blockIdx.x * blockDim.x + threadIdx.x;
  if (e >= N_EDGES) return;
  int dst = ei[N_EDGES + e];
  int rt  = et[e];
  atomicAdd(&deg[(size_t)dst * N_REL + rt], 1.0f);
}

// ---------------------------------------------------------------- edge scatter: S[dst][rel][f] += x_src[f]
__global__ void k_scatter(const float* __restrict__ ne, const int* __restrict__ ei,
                          const int* __restrict__ et, float* __restrict__ S) {
  int wave = threadIdx.x >> 5, lane = threadIdx.x & 31;
  int e = blockIdx.x * 8 + wave;          // one wave32 per edge
  if (e >= N_EDGES) return;
  int src = ei[e], dst = ei[N_EDGES + e], rt = et[e];
  const float4* x = (const float4*)(ne + (size_t)src * HID);
  float* s = S + ((size_t)dst * N_REL + rt) * HID;
#pragma unroll
  for (int i = 0; i < 6; ++i) {           // 32 lanes * 6 * float4 = 768
    int c4 = lane + 32 * i;
    float4 v = x[c4];
    atomicAdd(&s[c4 * 4 + 0], v.x);
    atomicAdd(&s[c4 * 4 + 1], v.y);
    atomicAdd(&s[c4 * 4 + 2], v.z);
    atomicAdd(&s[c4 * 4 + 3], v.w);
  }
}

// ---------------------------------------------------------------- fold comp & degree: A = [embeds | g_0..g_7] (bf16)
__global__ void k_combine(const float* __restrict__ S, const float* __restrict__ deg,
                          const float* __restrict__ comp, const float* __restrict__ ne,
                          u16* __restrict__ A) {
  __shared__ float sInv[N_REL];
  __shared__ float sComp[N_REL * N_BASES];
  int n = blockIdx.x, tid = threadIdx.x;
  if (tid < N_REL * N_BASES) sComp[tid] = comp[tid];
  if (tid < N_REL) {
    float d = (n < N_ENTITY) ? deg[(size_t)n * N_REL + tid] : 0.0f;
    sInv[tid] = 1.0f / fmaxf(d, 1.0f);    // PyG mean with clip(deg,1)
  }
  __syncthreads();
  u16* arow = A + (size_t)n * KTOT;
  for (int f = tid; f < HID; f += 256) {
    float emb = (n < N_ENTITY) ? ne[(size_t)n * HID + f] : 0.0f;
    arow[f] = f2bf(emb);
    float acc[N_BASES];
#pragma unroll
    for (int b = 0; b < N_BASES; ++b) acc[b] = 0.f;
    if (n < N_ENTITY) {
      const float* srow = S + (size_t)n * N_REL * HID + f;
#pragma unroll 4
      for (int r = 0; r < N_REL; ++r) {
        float v = srow[(size_t)r * HID] * sInv[r];
#pragma unroll
        for (int b = 0; b < N_BASES; ++b) acc[b] += sComp[r * N_BASES + b] * v;
      }
    }
#pragma unroll
    for (int b = 0; b < N_BASES; ++b) arow[HID + b * HID + f] = f2bf(acc[b]);
  }
}

// ---------------------------------------------------------------- weights -> bf16, pre-transposed (BT[n][k] = B[k][n])
__global__ void k_convW(const float* __restrict__ root, const float* __restrict__ basis,
                        const float* __restrict__ amat,
                        u16* __restrict__ BT, u16* __restrict__ AT) {
  size_t idx = (size_t)blockIdx.x * blockDim.x + threadIdx.x;
  const size_t NB1 = (size_t)HID * KTOT;
  const size_t NB2 = (size_t)HID * HID;
  if (idx < NB1) {
    int n = (int)(idx / KTOT), k = (int)(idx % KTOT);
    float v = (k < HID) ? root[(size_t)k * HID + n]
                        : basis[(size_t)(k - HID) * HID + n];
    BT[idx] = f2bf(v);
  } else if (idx < NB1 + NB2) {
    size_t j = idx - NB1;
    int n = (int)(j / HID), k = (int)(j % HID);
    AT[j] = f2bf(amat[(size_t)k * HID + n]);
  }
}

// ---------------------------------------------------------------- bf16 WMMA GEMM: C(M,768) = A(M,K) * BT(768,K)^T
// block tile 64x256, 8 waves of 32x64 (2x4 v_wmma_f32_16x16x32_bf16 frags), K step 32.
// Double-buffered LDS staged by global_load_async_to_lds_b128 (ASYNCcnt).
// mode 0: C = acc + bias[col] + add[row][col]   (ent = rgcn + embeds)
// mode 1: C = tanh(acc)                         (attention pre-activation)
__global__ __launch_bounds__(256)
void k_gemm_bf16(const u16* __restrict__ A, const u16* __restrict__ BT,
                 float* __restrict__ C, int M, int K, int mode,
                 const float* __restrict__ bias, const float* __restrict__ add) {
  __shared__ u16 As[2][64 * 32];      // [m][k] row-major            (4 KB x2)
  __shared__ u16 Bs[2][256 * 32];     // [n][k] (pre-transposed)    (16 KB x2)
  const int tid  = threadIdx.x;
  const int wave = tid >> 5, lane = tid & 31;
  const int msub = wave >> 2, nsub = wave & 3;          // 2 x 4 wave grid
  const int lm = lane & 15, hi = lane >> 4;
  const int m0 = blockIdx.x * 64, n0 = blockIdx.y * 256;
  const int aRow = tid >> 2, aK = (tid & 3) * 8;        // A: 8 u16 per thread
  const u16* gA = A + (size_t)(m0 + aRow) * K + aK;     // B: 32 u16 per thread (col = tid)
  const u16* gB = BT + (size_t)(n0 + tid) * K;
  v8f acc[2][4] = {};
  const int nk = K >> 5;

  // preload tile 0 into buffer 0
  {
    async_ld_b128(gA, &As[0][aRow * 32 + aK]);
#pragma unroll
    for (int c = 0; c < 4; ++c)
      async_ld_b128(gB + c * 8, &Bs[0][tid * 32 + c * 8]);
    wait_async0();
    __syncthreads();
  }

  for (int kt = 0; kt < nk; ++kt) {
    const int cur = kt & 1;
    if (kt + 1 < nk) {                   // DMA next tile into the other buffer
      const int nxt = cur ^ 1;
      const int ko = (kt + 1) * 32;
      async_ld_b128(gA + ko, &As[nxt][aRow * 32 + aK]);
#pragma unroll
      for (int c = 0; c < 4; ++c)
        async_ld_b128(gB + ko + c * 8, &Bs[nxt][tid * 32 + c * 8]);
    }
    // A frag (16x32 bf16): lanes 0-15 hold K0-7,K16-23; lanes 16-31 hold K8-15,K24-31
    Frag32B afr[2], bfr[4];
#pragma unroll
    for (int r = 0; r < 2; ++r) {
      const u16* p = &As[cur][(msub * 32 + r * 16 + lm) * 32 + hi * 8];
      afr[r].q[0] = *(const u32x4*)(p);
      afr[r].q[1] = *(const u32x4*)(p + 16);
    }
    // B frag (32x16 bf16): lanes 0-15 (col) hold K0-7,K16-23; lanes 16-31 hold K8-15,K24-31
#pragma unroll
    for (int c = 0; c < 4; ++c) {
      const u16* p = &Bs[cur][(nsub * 64 + c * 16 + lm) * 32 + hi * 8];
      bfr[c].q[0] = *(const u32x4*)(p);
      bfr[c].q[1] = *(const u32x4*)(p + 16);
    }
#pragma unroll
    for (int r = 0; r < 2; ++r)
#pragma unroll
      for (int c = 0; c < 4; ++c)
        acc[r][c] = __builtin_amdgcn_wmma_f32_16x16x32_bf16(
            false, afr[r].v, false, bfr[c].v, (short)0, acc[r][c], false, false);
    if (kt + 1 < nk) wait_async0();      // next tile committed to LDS
    __syncthreads();                     // everyone done reading cur; next tile visible
  }

  // C/D layout: VGPR v -> M = v + (lane<16 ? 0 : 8), N = lane%16
#pragma unroll
  for (int r = 0; r < 2; ++r)
#pragma unroll
    for (int c = 0; c < 4; ++c) {
      const int col = n0 + nsub * 64 + c * 16 + lm;
#pragma unroll
      for (int v = 0; v < 8; ++v) {
        const int row = m0 + msub * 32 + r * 16 + hi * 8 + v;
        float val = acc[r][c][v];
        if (mode == 0) {
          val += bias[col];
          if (row < N_ENTITY) val += add[(size_t)row * HID + col];
        } else {
          val = tanhf(val);
        }
        C[(size_t)row * HID + col] = val;
      }
    }
}

// ---------------------------------------------------------------- gather ent[entity_ids] (fp32 + bf16 copies)
__global__ void k_gather(const float* __restrict__ ENT, const int* __restrict__ ids,
                         float* __restrict__ HgF, u16* __restrict__ HgB) {
  size_t idx = (size_t)blockIdx.x * blockDim.x + threadIdx.x;
  if (idx >= (size_t)MB2 * HID) return;
  int row = (int)(idx / HID), f = (int)(idx % HID);
  float v = ENT[(size_t)ids[row] * HID + f];
  HgF[idx] = v;
  HgB[idx] = f2bf(v);
}

// ---------------------------------------------------------------- e = tanh(h@a)@b (tanh fused in GEMM), masked softmax, weighted sum
__global__ __launch_bounds__(256)
void k_attn(const float* __restrict__ T, const float* __restrict__ HgF,
            const float* __restrict__ bvec, const int* __restrict__ ids,
            float* __restrict__ out) {
  __shared__ float sb[HID];
  __shared__ float se[SEQ];
  __shared__ float sw[SEQ];
  const int b = blockIdx.x, tid = threadIdx.x;
  const int wave = tid >> 5, lane = tid & 31;
  for (int f = tid; f < HID; f += 256) sb[f] = bvec[f];
  __syncthreads();
  for (int l = wave; l < SEQ; l += 8) {
    const float* tr = T + ((size_t)b * SEQ + l) * HID;
    float acc = 0.f;
    for (int f = lane; f < HID; f += 32) acc += tr[f] * sb[f];
#pragma unroll
    for (int off = 16; off > 0; off >>= 1) acc += __shfl_xor(acc, off, 32);
    if (lane == 0) se[l] = acc;
  }
  __syncthreads();
  if (tid == 0) {
    int np = 0;
    for (int l = 0; l < SEQ; ++l) np += (ids[b * SEQ + l] != PAD_ID) ? 1 : 0;
    float bm = (np > 0) ? 1.0f : 0.0f;    // row-all-pad -> mask disabled (matches reference)
    float mx = -3.4e38f;
    for (int l = 0; l < SEQ; ++l) {
      float e = se[l];
      if (ids[b * SEQ + l] == PAD_ID) e += -1e30f * bm;
      se[l] = e;
      mx = fmaxf(mx, e);
    }
    float s = 0.f;
    for (int l = 0; l < SEQ; ++l) { float p = __expf(se[l] - mx); sw[l] = p; s += p; }
    float inv = 1.0f / s;
    for (int l = 0; l < SEQ; ++l) sw[l] *= inv;
  }
  __syncthreads();
  for (int f = tid; f < HID; f += 256) {
    float acc = 0.f;
    for (int l = 0; l < SEQ; ++l) acc += sw[l] * HgF[((size_t)b * SEQ + l) * HID + f];
    out[(size_t)b * HID + f] = acc;
  }
}

// ---------------------------------------------------------------- launcher
extern "C" void kernel_launch(void* const* d_in, const int* in_sizes, int n_in,
                              void* d_out, int out_size, void* d_ws, size_t ws_size,
                              hipStream_t stream) {
  const float* ne    = (const float*)d_in[0];
  const float* comp  = (const float*)d_in[1];
  const float* basis = (const float*)d_in[2];
  const float* root  = (const float*)d_in[3];
  const float* bias  = (const float*)d_in[4];
  const float* amat  = (const float*)d_in[5];
  const float* bvec  = (const float*)d_in[6];
  const int*   ei    = (const int*)d_in[7];
  const int*   et    = (const int*)d_in[8];
  const int*   ids   = (const int*)d_in[9];
  float* out = (float*)d_out;
  char* ws = (char*)d_ws;
  (void)in_sizes; (void)n_in; (void)out_size; (void)ws_size;

  const size_t S_BYTES   = (size_t)NPAD * N_REL * HID * sizeof(float);   // ~2.14 GiB
  const size_t A_BYTES   = (size_t)NPAD * KTOT * sizeof(u16);            // ~411 MiB
  const size_t BT_BYTES  = (size_t)HID * KTOT * sizeof(u16);
  const size_t ENT_BYTES = (size_t)NPAD * HID * sizeof(float);
  const size_t HGF_BYTES = (size_t)MB2 * HID * sizeof(float);
  const size_t HGB_BYTES = (size_t)MB2 * HID * sizeof(u16);
  const size_t AT_BYTES  = (size_t)HID * HID * sizeof(u16);

  float* S   = (float*)(ws);
  u16*   A   = (u16*)(ws + S_BYTES);
  u16*   BT  = (u16*)(ws + S_BYTES + A_BYTES);
  float* deg = (float*)(ws + S_BYTES + A_BYTES + BT_BYTES);
  // late-phase buffers alias the S region (S dead after k_combine)
  float* ENT = (float*)(ws);
  float* HgF = (float*)(ws + ENT_BYTES);
  u16*   HgB = (u16*)(ws + ENT_BYTES + HGF_BYTES);
  u16*   AT  = (u16*)(ws + ENT_BYTES + HGF_BYTES + HGB_BYTES);
  float* T   = (float*)(ws + ENT_BYTES + HGF_BYTES + HGB_BYTES + AT_BYTES);

  k_zero_f32<<<4096, 256, 0, stream>>>(S,   ((size_t)NPAD * N_REL * HID) / 4);
  k_zero_f32<<<1024, 256, 0, stream>>>(deg, ((size_t)NPAD * N_REL) / 4);
  k_deg<<<(N_EDGES + 255) / 256, 256, 0, stream>>>(ei, et, deg);
  k_scatter<<<N_EDGES / 8, 256, 0, stream>>>(ne, ei, et, S);
  k_combine<<<NPAD, 256, 0, stream>>>(S, deg, comp, ne, A);
  {
    size_t tot = (size_t)HID * KTOT + (size_t)HID * HID;
    k_convW<<<(unsigned)((tot + 255) / 256), 256, 0, stream>>>(root, basis, amat, BT, AT);
  }
  k_gemm_bf16<<<dim3(NPAD / 64, HID / 256), 256, 0, stream>>>(A, BT, ENT, NPAD, KTOT, 0, bias, ne);
  k_gather<<<(unsigned)(((size_t)MB2 * HID + 255) / 256), 256, 0, stream>>>(ENT, ids, HgF, HgB);
  k_gemm_bf16<<<dim3(MB2 / 64, HID / 256), 256, 0, stream>>>(HgB, AT, T, MB2, HID, 1, bias, ne);
  k_attn<<<BATCH, 256, 0, stream>>>(T, HgF, bvec, ids, out);
}
